// LSTM_Kmers_15848429322980
// MI455X (gfx1250) — matrix operand order repaired and checked by reading, not compiled, not run on hardware
//
#include <hip/hip_runtime.h>
#include <hip/hip_bf16.h>
#include <math.h>

// ---------------- problem constants (from reference) ----------------
#define S_LEN 4096
#define KMER  4
#define T_CH  (S_LEN * KMER)   // 16384 char steps
#define EC    64
#define HR    256
#define GR    (4 * HR)         // 1024 char gates
#define DL    128
#define EW    128
#define HW    512
#define GW    (4 * HW)         // 2048 word gates
#define KIN_W (EW + DL)        // 256 word-LSTM input dim

// scan partitioning
#define CH_NWG 8
#define CH_UPW (HR / CH_NWG)   // 32 units per WG
#define W_NWG  16
#define W_UPW  (HW / W_NWG)    // 32 units per WG

typedef __attribute__((ext_vector_type(16))) __bf16 v16bf;
typedef __attribute__((ext_vector_type(8)))  __bf16 v8bf;
typedef __attribute__((ext_vector_type(8)))  float  v8f;

static __device__ __forceinline__ float sigmoidf_(float x) {
    return 1.0f / (1.0f + __expf(-x));
}

static __device__ __forceinline__ unsigned lds_off_u32(const void* p) {
    // flat LDS aperture address: low 32 bits are the LDS byte offset
    return (unsigned)(unsigned long long)p;
}

// Async global->LDS broadcast of 512 bytes (16 B per lane) by wave 0.
// Per ISA 10.x: each lane copies 16 B from global into LDS; tracked by ASYNCcnt.
static __device__ __forceinline__ void async_bcast_to_lds(void* lds_dst,
                                                          const void* gsrc,
                                                          int lane) {
    unsigned l0 = lds_off_u32(lds_dst) + (unsigned)lane * 16u;
    unsigned long long g0 = (unsigned long long)gsrc + (unsigned long long)lane * 16ull;
    asm volatile("global_load_async_to_lds_b128 %0, %1, off"
                 :: "v"(l0), "v"(g0) : "memory");
}

// ---------------------------------------------------------------
// Pack weight W [N x Kd] (row-major) into the per-lane-contiguous B
// layout for v_wmma_f32_16x16x32_bf16.
// ---------------------------------------------------------------
__global__ void pack_B_bf16(const float* __restrict__ W, __bf16* __restrict__ Bp,
                            int N, int Kd) {
    int idx = blockIdx.x * blockDim.x + threadIdx.x;
    if (idx >= N * Kd) return;
    int n = idx / Kd, k = idx % Kd;
    int ntile = n >> 4, col = n & 15;
    int kc = k >> 5,  kk = k & 31;
    int lane = col + ((kk >= 16) ? 16 : 0);
    int KC = Kd >> 5;
    int dst = ((ntile * KC + kc) * 32 + lane) * 16 + (kk & 15);
    Bp[dst] = (__bf16)W[idx];
}

__global__ void gather_chars(const float* __restrict__ E_char,
                             const int* __restrict__ char_ids,
                             __bf16* __restrict__ Xc) {
    int idx = blockIdx.x * blockDim.x + threadIdx.x;
    if (idx >= T_CH * EC) return;
    int t = idx >> 6, e = idx & 63;
    Xc[idx] = (__bf16)E_char[char_ids[t] * EC + e];
}

__global__ void gather_words(const float* __restrict__ E_word,
                             const int* __restrict__ word_ids,
                             __bf16* __restrict__ XwIn) {
    int idx = blockIdx.x * blockDim.x + threadIdx.x;
    if (idx >= S_LEN * EW) return;
    int s = idx >> 7, e = idx & 127;
    XwIn[s * KIN_W + e] = (__bf16)E_word[word_ids[s] * EW + e];
}

__global__ void init_ctrl(unsigned* __restrict__ bars) {
    if (threadIdx.x < 8) bars[threadIdx.x] = 0u;
}

// ---------------------------------------------------------------
// Tiled GEMM C = A * B + bias via WMMA bf16->f32 (one wave per tile).
// mode 0: Cf[m*N+col] = acc+bias
// mode 1: Cbf[m*ldo + off + col] = bf16(tanh(acc+bias))
// ---------------------------------------------------------------
__global__ void gemm_wmma_bf16(const __bf16* __restrict__ A,
                               const __bf16* __restrict__ Bp,
                               const float* __restrict__ bias,
                               float* __restrict__ Cf,
                               __bf16* __restrict__ Cbf,
                               int M, int N, int Kd,
                               int mode, int ldo, int off) {
    const int Mt = M >> 4, Nt = N >> 4, KC = Kd >> 5;
    int tile = blockIdx.x * (blockDim.x >> 5) + (threadIdx.x >> 5);
    if (tile >= Mt * Nt) return;           // wave-uniform: EXEC all-1 for WMMA
    int mt = tile / Nt, nt = tile % Nt;
    int lane = threadIdx.x & 31;
    int row  = mt * 16 + (lane & 15);
    int base = (lane < 16) ? 0 : 16;

    v8f acc = {};
    for (int kc = 0; kc < KC; ++kc) {
        const v8bf* ap = (const v8bf*)(A + (size_t)row * Kd + kc * 32 + base);
        v8bf alo = ap[0];
        v8bf ahi = ap[1];
        v16bf av = __builtin_shufflevector(alo, ahi,
                   0,1,2,3,4,5,6,7,8,9,10,11,12,13,14,15);
        const v8bf* bp = (const v8bf*)(Bp + (size_t)((nt * KC + kc) * 32 + lane) * 16);
        v8bf blo = bp[0];
        v8bf bhi = bp[1];
        v16bf bvv = __builtin_shufflevector(blo, bhi,
                   0,1,2,3,4,5,6,7,8,9,10,11,12,13,14,15);
        acc = __builtin_amdgcn_wmma_f32_16x16x32_bf16(
                  false, av, false, bvv, (short)0, acc, false, false);
    }

    int col  = nt * 16 + (lane & 15);
    float bb = bias[col];
    int mrow = mt * 16 + ((lane < 16) ? 0 : 8);
    if (mode == 0) {
#pragma unroll
        for (int v = 0; v < 8; ++v)
            Cf[(size_t)(mrow + v) * N + col] = acc[v] + bb;
    } else {
#pragma unroll
        for (int v = 0; v < 8; ++v)
            Cbf[(size_t)(mrow + v) * ldo + off + col] = (__bf16)tanhf(acc[v] + bb);
    }
}

// ---------------------------------------------------------------
// Char-LSTM scan across CH_NWG workgroups. WG p owns units
// [p*32, p*32+32): all 4 gate rows for those units (128 rows x 256).
// Per step: 1024 thr, 8 threads/row (32 MACs each) + shfl_xor reduce,
// h exchanged via L2 + atomic spin barrier + async global->LDS bcast.
// ---------------------------------------------------------------
__global__ void __launch_bounds__(1024)
char_lstm_scan(const float* __restrict__ Xr,      // [T_CH x GR]
               const float* __restrict__ Whh,     // [GR x HR]
               __bf16* __restrict__ HwordBF,      // [S_LEN x HR]
               float* __restrict__ hbuf,          // [2 x HR]
               unsigned* __restrict__ bar) {
    __shared__ float h_s[HR];
    __shared__ float zloc[4 * CH_UPW];
    const int tid  = threadIdx.x;
    const int p    = blockIdx.x;
    const int lr   = tid >> 3;             // local row 0..127
    const int sub  = tid & 7;              // k-slice 0..7 (32 k each)
    const int gate = lr >> 5;
    const int u    = lr & 31;
    const int gr   = gate * HR + p * CH_UPW + u;   // global gate row

    if (tid < HR) h_s[tid] = 0.0f;
    float cst = 0.0f;
    __syncthreads();

    const float4* wrow = (const float4*)(Whh + (size_t)gr * HR + sub * 32);
    const float4* h4   = (const float4*)h_s + sub * 8;

    for (int t = 0; t < T_CH; ++t) {
        float acc = (sub == 0) ? Xr[(size_t)t * GR + gr] : 0.0f;
        if (sub == 0 && t + 1 < T_CH)
            __builtin_prefetch(&Xr[(size_t)(t + 1) * GR + gr], 0, 3);
#pragma unroll
        for (int k = 0; k < 8; ++k) {
            float4 w = wrow[k];
            float4 h = h4[k];
            acc += w.x * h.x + w.y * h.y + w.z * h.z + w.w * h.w;
        }
        acc += __shfl_xor(acc, 1, 32);
        acc += __shfl_xor(acc, 2, 32);
        acc += __shfl_xor(acc, 4, 32);
        if (sub == 0) zloc[lr] = acc;
        __syncthreads();

        float* hnext = hbuf + ((t + 1) & 1) * HR;
        if (tid < CH_UPW) {
            float iz = sigmoidf_(zloc[tid]);
            float fz = sigmoidf_(zloc[CH_UPW + tid]);
            float gz = tanhf(zloc[2 * CH_UPW + tid]);
            float oz = sigmoidf_(zloc[3 * CH_UPW + tid]);
            cst = fz * cst + iz * gz;
            float hn = oz * tanhf(cst);
            hnext[p * CH_UPW + tid] = hn;
            if ((t & (KMER - 1)) == (KMER - 1))
                HwordBF[(size_t)(t >> 2) * HR + p * CH_UPW + tid] = (__bf16)hn;
            __threadfence();               // release h slice to agent scope
        }
        __syncthreads();

        if (t + 1 < T_CH) {
            if (tid == 0) {
                __hip_atomic_fetch_add(bar, 1u, __ATOMIC_RELEASE,
                                       __HIP_MEMORY_SCOPE_AGENT);
                unsigned target = (unsigned)CH_NWG * (unsigned)(t + 1);
                while (__hip_atomic_load(bar, __ATOMIC_ACQUIRE,
                                         __HIP_MEMORY_SCOPE_AGENT) < target)
                    __builtin_amdgcn_s_sleep(1);
            }
            __syncthreads();
            if (tid < 32) {                // wave 0: async h broadcast, 1 KB
                async_bcast_to_lds(h_s, hnext, tid);           // bytes 0..511
                asm volatile("global_load_async_to_lds_b128 %0, %1, off offset:512"
                             :: "v"(lds_off_u32(h_s) + (unsigned)tid * 16u),
                                "v"((unsigned long long)hnext + (unsigned long long)tid * 16ull)
                             : "memory");
                asm volatile("s_wait_asynccnt 0x0" ::: "memory");
            }
            __syncthreads();
        }
    }
}

// ---------------------------------------------------------------
// Word-LSTM scan across W_NWG workgroups (units split 32/WG,
// 128 rows x 512 dot per WG, 8 threads/row => 64 MACs each).
// ---------------------------------------------------------------
__global__ void __launch_bounds__(1024)
word_lstm_scan(const float* __restrict__ Xw,      // [S_LEN x GW]
               const float* __restrict__ Whh,     // [GW x HW]
               float* __restrict__ out,           // [S_LEN x HW]
               float* __restrict__ hbuf,          // [2 x HW]
               unsigned* __restrict__ bar) {
    __shared__ float h_s[HW];
    __shared__ float zloc[4 * W_UPW];
    const int tid  = threadIdx.x;
    const int p    = blockIdx.x;
    const int lr   = tid >> 3;
    const int sub  = tid & 7;              // k-slice 0..7 (64 k each)
    const int gate = lr >> 5;
    const int u    = lr & 31;
    const int gr   = gate * HW + p * W_UPW + u;

    if (tid < HW) h_s[tid] = 0.0f;
    float cst = 0.0f;
    __syncthreads();

    const float4* wrow = (const float4*)(Whh + (size_t)gr * HW + sub * 64);
    const float4* h4   = (const float4*)h_s + sub * 16;

    for (int t = 0; t < S_LEN; ++t) {
        float acc = (sub == 0) ? Xw[(size_t)t * GW + gr] : 0.0f;
        if (sub == 0 && t + 1 < S_LEN)
            __builtin_prefetch(&Xw[(size_t)(t + 1) * GW + gr], 0, 3);
#pragma unroll
        for (int k = 0; k < 16; ++k) {
            float4 w = wrow[k];
            float4 h = h4[k];
            acc += w.x * h.x + w.y * h.y + w.z * h.z + w.w * h.w;
        }
        acc += __shfl_xor(acc, 1, 32);
        acc += __shfl_xor(acc, 2, 32);
        acc += __shfl_xor(acc, 4, 32);
        if (sub == 0) zloc[lr] = acc;
        __syncthreads();

        float* hnext = hbuf + ((t + 1) & 1) * HW;
        if (tid < W_UPW) {
            float iz = sigmoidf_(zloc[tid]);
            float fz = sigmoidf_(zloc[W_UPW + tid]);
            float gz = tanhf(zloc[2 * W_UPW + tid]);
            float oz = sigmoidf_(zloc[3 * W_UPW + tid]);
            cst = fz * cst + iz * gz;
            float hn = oz * tanhf(cst);
            hnext[p * W_UPW + tid] = hn;
            out[(size_t)t * HW + p * W_UPW + tid] = hn;
            __threadfence();
        }
        __syncthreads();

        if (t + 1 < S_LEN) {
            if (tid == 0) {
                __hip_atomic_fetch_add(bar, 1u, __ATOMIC_RELEASE,
                                       __HIP_MEMORY_SCOPE_AGENT);
                unsigned target = (unsigned)W_NWG * (unsigned)(t + 1);
                while (__hip_atomic_load(bar, __ATOMIC_ACQUIRE,
                                         __HIP_MEMORY_SCOPE_AGENT) < target)
                    __builtin_amdgcn_s_sleep(1);
            }
            __syncthreads();
            if (tid < 32) {                // wave 0: async h broadcast, 2 KB
                unsigned l0 = lds_off_u32(h_s) + (unsigned)tid * 16u;
                unsigned long long g0 = (unsigned long long)hnext
                                      + (unsigned long long)tid * 16ull;
                asm volatile("global_load_async_to_lds_b128 %0, %1, off"
                             :: "v"(l0), "v"(g0) : "memory");
                asm volatile("global_load_async_to_lds_b128 %0, %1, off offset:512"
                             :: "v"(l0), "v"(g0) : "memory");
                asm volatile("global_load_async_to_lds_b128 %0, %1, off offset:1024"
                             :: "v"(l0), "v"(g0) : "memory");
                asm volatile("global_load_async_to_lds_b128 %0, %1, off offset:1536"
                             :: "v"(l0), "v"(g0) : "memory");
                asm volatile("s_wait_asynccnt 0x0" ::: "memory");
            }
            __syncthreads();
        }
    }
}

// ---------------------------------------------------------------
extern "C" void kernel_launch(void* const* d_in, const int* in_sizes, int n_in,
                              void* d_out, int out_size, void* d_ws, size_t ws_size,
                              hipStream_t stream) {
    const float* E_char = (const float*)d_in[0];
    const float* W_ih_r = (const float*)d_in[1];
    const float* W_hh_r = (const float*)d_in[2];
    const float* b_r    = (const float*)d_in[3];
    const float* W_lat  = (const float*)d_in[4];
    const float* b_lat  = (const float*)d_in[5];
    const float* E_word = (const float*)d_in[6];
    const float* W_ih_w = (const float*)d_in[7];
    const float* W_hh_w = (const float*)d_in[8];
    const float* b_w    = (const float*)d_in[9];
    const int* word_ids = (const int*)d_in[10];
    const int* char_ids = (const int*)d_in[11];
    float* out = (float*)d_out;

    char* ws = (char*)d_ws;
    size_t off = 0;
    float*  Xr      = (float*)(ws + off);  off += (size_t)T_CH  * GR    * 4;
    float*  Xw      = (float*)(ws + off);  off += (size_t)S_LEN * GW    * 4;
    __bf16* XcBF    = (__bf16*)(ws + off); off += (size_t)T_CH  * EC    * 2;
    __bf16* XwInBF  = (__bf16*)(ws + off); off += (size_t)S_LEN * KIN_W * 2;
    __bf16* HwordBF = (__bf16*)(ws + off); off += (size_t)S_LEN * HR    * 2;
    __bf16* WihrP   = (__bf16*)(ws + off); off += (size_t)GR * EC       * 2;
    __bf16* WlatP   = (__bf16*)(ws + off); off += (size_t)DL * HR       * 2;
    __bf16* WihwP   = (__bf16*)(ws + off); off += (size_t)GW * KIN_W    * 2;
    unsigned* bars  = (unsigned*)(ws + off); off += 1024;   // [0]=char, [1]=word
    float*  hbufC   = (float*)(ws + off);  off += 2 * HR * 4;
    float*  hbufW   = (float*)(ws + off);  off += 2 * HW * 4;
    (void)off; (void)ws_size; (void)in_sizes; (void)n_in; (void)out_size;

    // stage 0: pack weights + gather embeddings + zero barrier counters
    pack_B_bf16<<<(GR * EC + 255) / 256, 256, 0, stream>>>(W_ih_r, WihrP, GR, EC);
    pack_B_bf16<<<(DL * HR + 255) / 256, 256, 0, stream>>>(W_lat, WlatP, DL, HR);
    pack_B_bf16<<<(GW * KIN_W + 255) / 256, 256, 0, stream>>>(W_ih_w, WihwP, GW, KIN_W);
    gather_chars<<<(T_CH * EC + 255) / 256, 256, 0, stream>>>(E_char, char_ids, XcBF);
    gather_words<<<(S_LEN * EW + 255) / 256, 256, 0, stream>>>(E_word, word_ids, XwInBF);
    init_ctrl<<<1, 64, 0, stream>>>(bars);

    // stage 1: Xr = Xc @ W_ih_r^T + b_r   (WMMA)
    {
        int tiles = (T_CH / 16) * (GR / 16);
        gemm_wmma_bf16<<<(tiles + 7) / 8, 256, 0, stream>>>(
            XcBF, WihrP, b_r, Xr, nullptr, T_CH, GR, EC, 0, 0, 0);
    }

    // stage 2: char-LSTM scan (8 WGs, unit-partitioned)
    char_lstm_scan<<<CH_NWG, 1024, 0, stream>>>(Xr, W_hh_r, HwordBF, hbufC, &bars[0]);

    // stage 3: latent = tanh(h_word @ W_lat^T + b_lat) -> bf16 into XwIn[:,128:]
    {
        int tiles = (S_LEN / 16) * (DL / 16);
        gemm_wmma_bf16<<<(tiles + 7) / 8, 256, 0, stream>>>(
            HwordBF, WlatP, b_lat, nullptr, XwInBF, S_LEN, DL, HR, 1, KIN_W, EW);
    }

    // stage 4: Xw = XwIn @ W_ih_w^T + b_w   (WMMA)
    {
        int tiles = (S_LEN / 16) * (GW / 16);
        gemm_wmma_bf16<<<(tiles + 7) / 8, 256, 0, stream>>>(
            XwInBF, WihwP, b_w, Xw, nullptr, S_LEN, GW, KIN_W, 0, 0, 0);
    }

    // stage 5: word-LSTM scan (16 WGs) -> d_out
    word_lstm_scan<<<W_NWG, 1024, 0, stream>>>(Xw, W_hh_w, out, hbufW, &bars[1]);
}
